// ConditionalConv_70300024701395
// MI455X (gfx1250) — matrix operand order repaired and checked by reading, not compiled, not run on hardware
//
#include <hip/hip_runtime.h>

// ---------------------------------------------------------------------------
// ConditionalConv (hypernetwork MLP -> per-sample 3x3 conv) for MI455X/gfx1250
//
// Roofline: W2 stream (170 MB f32) + x/out (128 MB) ~= 300 MB -> ~13 us floor
// at 23.3 TB/s. FLOPs only 12.4 G -> memory bound. All three stages run as
// v_wmma_f32_16x16x32_f16 (f16 in, f32 accumulate). The dominant W2 stream is
// K-split 4 ways (deterministic partial buffers, no float atomics) so enough
// waves are in flight to cover HBM latency; a tiny reduce kernel folds the
// partials + bias into f16 per-sample weight matrices. x (64 MB) fits the
// 192 MB L2, so the conv's 3-row halo re-reads stay on-chip.
// ---------------------------------------------------------------------------

typedef __attribute__((ext_vector_type(16))) _Float16 v16h;
typedef __attribute__((ext_vector_type(8)))  float    v8f;

#define B_       32
#define CIN_     32
#define COUT_    32
#define H_       128
#define W_       128
#define COND_    256
#define HIDDEN_  4608
#define PARAMS_  9216
#define KCONV_   288            // CIN * 3 * 3
#define KSPLIT_  4
#define KCHUNK_  (HIDDEN_ / KSPLIT_)   // 1152, multiple of 32

// D = A*B + C, 16x16x32, f16 inputs, f32 accum (codegen-verified builtin)
static __device__ __forceinline__ v8f wmma16(v16h a, v16h b, v8f c) {
  return __builtin_amdgcn_wmma_f32_16x16x32_f16(
      /*neg_a=*/false, a, /*neg_b=*/false, b,
      /*c_mod=*/(short)0, c, /*reuse_a=*/false, /*reuse_b=*/false);
}

// A-matrix fragment (16x32, 16-bit): lane m = row; element e -> K=hi*16+e.
// Two contiguous 8-half (16B) runs per lane.
static __device__ __forceinline__ v16h a_frag_f32(const float* __restrict__ row,
                                                  int kc, int hi) {
  v16h a;
  const float* r0 = row + kc + hi * 16;
  const float* r1 = row + kc + hi * 16 + 8;
#pragma unroll
  for (int i = 0; i < 8; ++i) {
    a[i]     = (_Float16)r0[i];
    a[8 + i] = (_Float16)r1[i];
  }
  return a;
}

static __device__ __forceinline__ v16h a_frag_f16(const _Float16* __restrict__ row,
                                                  int kc, int hi) {
  v16h a;
  const _Float16* r0 = row + kc + hi * 16;
  const _Float16* r1 = row + kc + hi * 16 + 8;
#pragma unroll
  for (int i = 0; i < 8; ++i) {
    a[i]     = r0[i];
    a[8 + i] = r1[i];
  }
  return a;
}

// B-matrix fragment (32x16, 16-bit): lane col n = lane%16; element e -> K=hi*16+e.
// Source is one logical B column laid out contiguously along K (f32).
static __device__ __forceinline__ v16h b_frag_f32(const float* __restrict__ col,
                                                  int kc, int hi) {
  v16h b;
  const float* r = col + kc + hi * 16;
#pragma unroll
  for (int i = 0; i < 16; ++i) b[i] = (_Float16)r[i];
  return b;
}

// ---------------------------------------------------------------------------
// Kernel 1: hOut[b][m] = relu(sum_k W1[m][k] * cond[b][k] + b1[m]), f16 out.
// M = HIDDEN (A rows = W1 rows, contiguous K), N = batch (2 tiles), K = 256.
// Grid: 4608/128 = 36 blocks x 8 waves x 16 rows. Tiny kernel (75 MFLOP).
// ---------------------------------------------------------------------------
__global__ void k_mlp1(const float* __restrict__ cond, const float* __restrict__ W1,
                       const float* __restrict__ b1, _Float16* __restrict__ hOut) {
  const int lane  = threadIdx.x & 31;
  const int wave  = threadIdx.x >> 5;
  const int mlane = lane & 15;
  const int hi    = lane >> 4;
  const int m0    = blockIdx.x * 128 + wave * 16;

  const float* w1row = W1 + (size_t)(m0 + mlane) * COND_;
  const float* c0    = cond + (size_t)mlane * COND_;          // batch 0..15
  const float* c1    = cond + (size_t)(16 + mlane) * COND_;   // batch 16..31

  v8f acc0 = {}, acc1 = {};
#pragma unroll
  for (int kc = 0; kc < COND_; kc += 32) {
    v16h a   = a_frag_f32(w1row, kc, hi);
    v16h bf0 = b_frag_f32(c0, kc, hi);
    v16h bf1 = b_frag_f32(c1, kc, hi);
    acc0 = wmma16(a, bf0, acc0);
    acc1 = wmma16(a, bf1, acc1);
  }
  // D element r: M = r + 8*hi (tile-relative), N = lane%16
#pragma unroll
  for (int r = 0; r < 8; ++r) {
    const int m = m0 + r + 8 * hi;
    const float bias = b1[m];
    float v0 = acc0[r] + bias; v0 = v0 > 0.f ? v0 : 0.f;
    float v1 = acc1[r] + bias; v1 = v1 > 0.f ? v1 : 0.f;
    hOut[(size_t)mlane * HIDDEN_ + m]        = (_Float16)v0;
    hOut[(size_t)(16 + mlane) * HIDDEN_ + m] = (_Float16)v1;
  }
}

// ---------------------------------------------------------------------------
// Kernel 2: partial[ks][b][p] = sum_{k in split ks} h[b][k] * W2[p][k].
// GEMM view: M = batch (A = h f16, L2-resident), N = p (B rows = W2 rows;
// lanes L and L+16 stream adjacent 64B halves of the same W2 row -> full
// 128B-line utilization), K split 4 ways for occupancy: 72x4 blocks x 8 waves
// = 2304 waves streaming the 170 MB of W2 exactly once, prefetching ahead.
// ---------------------------------------------------------------------------
__global__ void k_mlp2(const _Float16* __restrict__ h, const float* __restrict__ W2,
                       float* __restrict__ wPart) {
  const int lane  = threadIdx.x & 31;
  const int wave  = threadIdx.x >> 5;
  const int mlane = lane & 15;
  const int hi    = lane >> 4;
  const int ntile = blockIdx.x * 8 + wave;
  const int ks    = blockIdx.y;
  const int p     = ntile * 16 + mlane;
  const int kbase = ks * KCHUNK_;

  const float*    w2row = W2 + (size_t)p * HIDDEN_;
  const _Float16* hrow0 = h + (size_t)mlane * HIDDEN_;
  const _Float16* hrow1 = h + (size_t)(16 + mlane) * HIDDEN_;

  v8f acc0 = {}, acc1 = {};
  for (int kc = kbase; kc < kbase + KCHUNK_; kc += 32) {
    __builtin_prefetch(w2row + kc + 1024, 0, 1);   // global_prefetch_b8
    v16h bf = b_frag_f32(w2row, kc, hi);
    v16h a0 = a_frag_f16(hrow0, kc, hi);           // batch 0..15
    v16h a1 = a_frag_f16(hrow1, kc, hi);           // batch 16..31
    acc0 = wmma16(a0, bf, acc0);
    acc1 = wmma16(a1, bf, acc1);
  }

  float* part = wPart + (size_t)ks * (B_ * PARAMS_);
#pragma unroll
  for (int r = 0; r < 8; ++r) {
    const int bIdx0 = r + 8 * hi;
    const int bIdx1 = 16 + r + 8 * hi;
    part[(size_t)bIdx0 * PARAMS_ + p] = acc0[r];
    part[(size_t)bIdx1 * PARAMS_ + p] = acc1[r];
  }
}

// ---------------------------------------------------------------------------
// Kernel 2b: fold K-split partials + bias -> f16 per-sample weight matrices.
// w16[b][p] = f16( b2[p] + sum_ks partial[ks][b][p] ).  294912 elems, 4/thread.
// ---------------------------------------------------------------------------
__global__ void k_reduce(const float* __restrict__ wPart, const float* __restrict__ b2,
                         _Float16* __restrict__ w16) {
  const int base = (blockIdx.x * 256 + threadIdx.x) * 4;
#pragma unroll
  for (int j = 0; j < 4; ++j) {
    const int i = base + j;
    float s = b2[i % PARAMS_];
#pragma unroll
    for (int ks = 0; ks < KSPLIT_; ++ks)
      s += wPart[(size_t)ks * (B_ * PARAMS_) + i];
    w16[i] = (_Float16)s;
  }
}

// ---------------------------------------------------------------------------
// Kernel 3: per-sample 3x3 conv as implicit GEMM.
// Block = (sample b, output row y), 8 waves. M = Cout (2 tiles), N = 16-pixel
// tiles (8 waves cover the 128-wide row), K = CIN*9 = 288.
// Input halo tile staged f16 in LDS: [ci=32][3 rows][132 cols] = 24.75 KB.
// Weights read from the f16 workspace (18 KB/sample -> cache resident).
// ---------------------------------------------------------------------------
__global__ void k_conv(const float* __restrict__ x, const _Float16* __restrict__ wAll,
                       float* __restrict__ out) {
  __shared__ _Float16 xs[CIN_][3][132];

  const int y = blockIdx.x;
  const int b = blockIdx.y;

  // Cooperative halo load + f32->f16 convert (zero-padded borders).
  for (int t = threadIdx.x; t < CIN_ * 3 * 132; t += 256) {
    const int col = t % 132;
    const int tmp = t / 132;
    const int row = tmp % 3;
    const int ci  = tmp / 3;
    const int iy  = y + row - 1;
    const int ix  = col - 1;
    float v = 0.f;
    if (iy >= 0 && iy < H_ && ix >= 0 && ix < W_)
      v = x[(((size_t)b * CIN_ + ci) * H_ + iy) * W_ + ix];
    xs[ci][row][col] = (_Float16)v;
  }
  __syncthreads();

  const int lane  = threadIdx.x & 31;
  const int wave  = threadIdx.x >> 5;
  const int mlane = lane & 15;
  const int hi    = lane >> 4;
  const int px    = wave * 16 + mlane;     // this lane's output column (N)

  const _Float16* wb  = wAll + (size_t)b * PARAMS_;         // [32 co][288 k]
  const _Float16* wr0 = wb + (size_t)mlane * KCONV_;        // co 0..15
  const _Float16* wr1 = wb + (size_t)(16 + mlane) * KCONV_; // co 16..31

  v8f acc0 = {}, acc1 = {};
#pragma unroll
  for (int kc = 0; kc < KCONV_; kc += 32) {
    // B fragment: im2col gather from LDS. element e -> K = kc + hi*16 + e,
    // K decomposes as (ci, ky, kx) with k = ci*9 + ky*3 + kx.
    v16h bf;
    const int kbase = kc + hi * 16;
#pragma unroll
    for (int e = 0; e < 16; ++e) {
      const int k  = kbase + e;
      const int ci = k / 9;
      const int kk = k - ci * 9;
      const int ky = kk / 3;
      const int kx = kk - ky * 3;
      bf[e] = xs[ci][ky][px + kx];
    }
    v16h a0 = a_frag_f16(wr0, kc, hi);
    v16h a1 = a_frag_f16(wr1, kc, hi);
    acc0 = wmma16(a0, bf, acc0);
    acc1 = wmma16(a1, bf, acc1);
  }

#pragma unroll
  for (int r = 0; r < 8; ++r) {
    const int co0 = r + 8 * hi;
    const int co1 = 16 + r + 8 * hi;
    out[(((size_t)b * COUT_ + co0) * H_ + y) * W_ + px] = acc0[r];
    out[(((size_t)b * COUT_ + co1) * H_ + y) * W_ + px] = acc1[r];
  }
}

// ---------------------------------------------------------------------------
extern "C" void kernel_launch(void* const* d_in, const int* in_sizes, int n_in,
                              void* d_out, int out_size, void* d_ws, size_t ws_size,
                              hipStream_t stream) {
  (void)in_sizes; (void)n_in; (void)out_size; (void)ws_size;

  const float* x    = (const float*)d_in[0];   // [32][32][128][128]
  const float* cond = (const float*)d_in[1];   // [32][256]
  const float* W1   = (const float*)d_in[2];   // [4608][256]
  const float* b1   = (const float*)d_in[3];   // [4608]
  const float* W2   = (const float*)d_in[4];   // [9216][4608]
  const float* b2   = (const float*)d_in[5];   // [9216]
  float* out        = (float*)d_out;           // [32][32][128][128]

  // Workspace layout (all 256B-aligned):
  //   h16   : f16 [32][4608]                 = 294912 B
  //   wPart : f32 [KSPLIT][32][9216]          = 4718592 B
  //   w16   : f16 [32][9216]                  = 589824 B
  char* ws = (char*)d_ws;
  _Float16* h16   = (_Float16*)ws;
  float*    wPart = (float*)(ws + 294912);
  _Float16* w16   = (_Float16*)(ws + 294912 + 4718592);

  k_mlp1  <<<dim3(HIDDEN_ / 128),          dim3(256), 0, stream>>>(cond, W1, b1, h16);
  k_mlp2  <<<dim3(PARAMS_ / 128, KSPLIT_), dim3(256), 0, stream>>>(h16, W2, wPart);
  k_reduce<<<dim3(B_ * PARAMS_ / 1024),    dim3(256), 0, stream>>>(wPart, b2, w16);
  k_conv  <<<dim3(H_, B_),                 dim3(256), 0, stream>>>(x, w16, out);
}
